// GNN_62311385530802
// MI455X (gfx1250) — compile-verified, matrix operand order checked by reading, not run on hardware
//
#include <hip/hip_runtime.h>
#include <hip/hip_bf16.h>
#include <cmath>

#define N_NODES 100000
#define N_EDGES 800000
#define BGRAPH  64
#define IN_F    256
#define HC1C    128
#define HC2C    32
#define SAC     350
#define NEG_SLOPE 0.2f

typedef __attribute__((ext_vector_type(16))) _Float16 v16h;
typedef __attribute__((ext_vector_type(8)))  _Float16 v8h;
typedef __attribute__((ext_vector_type(4)))  _Float16 v4h;
typedef __attribute__((ext_vector_type(8)))  float    v8f;

// ---------------------------------------------------------------------------
// Convert f32 activations -> zero-padded f16 (row ld = kin_pad, kin_pad % 32 == 0).
// ---------------------------------------------------------------------------
__global__ void cvt_f16_pad_kernel(const float* __restrict__ X, _Float16* __restrict__ Xh,
                                   long nrows, int kin, int kin_pad)
{
    long i4 = (long)blockIdx.x * blockDim.x + threadIdx.x;
    const long total4 = nrows * (kin_pad >> 2);
    if (i4 >= total4) return;
    const long i = i4 * 4;
    const int  k = (int)(i % kin_pad);
    const long r = i / kin_pad;
    const float* xr = X + r * (long)kin + k;
    v4h o;
#pragma unroll
    for (int j = 0; j < 4; ++j)
        o[j] = (k + j < kin) ? (_Float16)xr[j] : (_Float16)0.0f;
    *(v4h*)(Xh + i) = o;
}

// ---------------------------------------------------------------------------
// Pack f32 weights (kin x kout, row-major) into WMMA B-fragment layout:
//   Wp[((tn*nkc + kc)*32 + lane)*16 + e]  (f16, zero-padded; ntn rounded up to 4)
// ---------------------------------------------------------------------------
__global__ void pack_w_kernel(const float* __restrict__ W, _Float16* __restrict__ Wp,
                              int kin, int kout, int ntn, int nkc)
{
    int i = blockIdx.x * blockDim.x + threadIdx.x;
    const int total = ntn * nkc * 32;
    if (i >= total) return;
    const int lane = i & 31;
    const int kc   = (i >> 5) % nkc;
    const int tn   = (i >> 5) / nkc;
    const int half = lane >> 4;
    const int l16  = lane & 15;
    const int col  = tn * 16 + l16;
    const int k0   = kc * 32;
    _Float16* out = Wp + ((long)(tn * nkc + kc) * 32 + lane) * 16;
#pragma unroll
    for (int e = 0; e < 8; ++e) {
        const int ka = k0 + half * 8 + e;
        const int kb = k0 + 16 + half * 8 + e;
        out[e]     = (col < kout && ka < kin) ? (_Float16)W[(long)ka * kout + col] : (_Float16)0.0f;
        out[e + 8] = (col < kout && kb < kin) ? (_Float16)W[(long)kb * kout + col] : (_Float16)0.0f;
    }
}

// ---------------------------------------------------------------------------
// WMMA GEMM: Y = act(Xh @ W + bias). One wave32 computes a 16x64 tile:
// one shared A fragment + 4 B fragments -> 4 v_wmma per k-step (10 b128 loads).
// nrows must be a multiple of 16; ntn (B tiles) is a multiple of 4, zero-padded.
// ---------------------------------------------------------------------------
__global__ void wmma_linear_kernel(const _Float16* __restrict__ Xh,
                                   const _Float16* __restrict__ Wp,
                                   const float* __restrict__ bias,
                                   float* __restrict__ Y,
                                   int nrows, int kin_pad, int kout, int nkc, int relu_act)
{
    const int lane = threadIdx.x;        // blockDim.x == 32
    const int half = lane >> 4;
    const int l16  = lane & 15;
    const int tm   = blockIdx.x;
    const int tn4  = blockIdx.y;         // group of 4 column tiles

    const _Float16* __restrict__ ap = Xh + (long)(tm * 16 + l16) * kin_pad + half * 8;
    const _Float16* __restrict__ bp = Wp + (long)(tn4 * 4) * nkc * 512 + lane * 16;
    const long tstride = (long)nkc * 512;

    v8f acc[4] = {{}, {}, {}, {}};
    for (int kc = 0; kc < nkc; ++kc) {
        const v8h alo = *(const v8h*)(ap + kc * 32);
        const v8h ahi = *(const v8h*)(ap + kc * 32 + 16);
        const v16h a = __builtin_shufflevector(alo, ahi,
                        0, 1, 2, 3, 4, 5, 6, 7, 8, 9, 10, 11, 12, 13, 14, 15);
        const long kb = (long)kc * 512;
        const v16h b0 = *(const v16h*)(bp + 0 * tstride + kb);
        const v16h b1 = *(const v16h*)(bp + 1 * tstride + kb);
        const v16h b2 = *(const v16h*)(bp + 2 * tstride + kb);
        const v16h b3 = *(const v16h*)(bp + 3 * tstride + kb);
        acc[0] = __builtin_amdgcn_wmma_f32_16x16x32_f16(false, a, false, b0, (short)0, acc[0], false, false);
        acc[1] = __builtin_amdgcn_wmma_f32_16x16x32_f16(false, a, false, b1, (short)0, acc[1], false, false);
        acc[2] = __builtin_amdgcn_wmma_f32_16x16x32_f16(false, a, false, b2, (short)0, acc[2], false, false);
        acc[3] = __builtin_amdgcn_wmma_f32_16x16x32_f16(false, a, false, b3, (short)0, acc[3], false, false);
    }
#pragma unroll
    for (int j = 0; j < 4; ++j) {
        const int col = (tn4 * 4 + j) * 16 + l16;
        if (col < kout) {
            const float bv = bias ? bias[col] : 0.0f;
#pragma unroll
            for (int r = 0; r < 8; ++r) {
                const int m = tm * 16 + r + half * 8;   // C/D: M = r + 8*half
                float v = acc[j][r] + bv;
                if (relu_act) v = fmaxf(v, 0.0f);
                Y[(long)m * kout + col] = v;
            }
        }
    }
}

// ---------------------------------------------------------------------------
// Small utility kernels
// ---------------------------------------------------------------------------
__global__ void fill_f32_kernel(float* __restrict__ p, float v, long n)
{
    long i = (long)blockIdx.x * blockDim.x + threadIdx.x;
    if (i < n) p[i] = v;
}

__global__ void init_bias_rows_kernel(float* __restrict__ out,
                                      const float* __restrict__ bias, long n, int d)
{
    long i = (long)blockIdx.x * blockDim.x + threadIdx.x;
    if (i < n) out[i] = bias[i % d];
}

__global__ void relu_ip_kernel(float* __restrict__ p, long n)
{
    long i = (long)blockIdx.x * blockDim.x + threadIdx.x;
    if (i < n) p[i] = fmaxf(p[i], 0.0f);
}

__device__ __forceinline__ void atomicMaxF(float* addr, float val)
{
    unsigned int* ua = (unsigned int*)addr;
    unsigned int cur = __float_as_uint(*addr);
    while (__uint_as_float(cur) < val) {
        unsigned int prev = atomicCAS(ua, cur, __float_as_uint(val));
        if (prev == cur) break;
        cur = prev;
    }
}

// ---------------------------------------------------------------------------
// GATv2 edge phase. 8 lanes cooperate per edge: coalesced 128B row segments.
// pass 1 — per-edge score (subwave dot + shfl reduction) + segment max
// ---------------------------------------------------------------------------
__global__ void edge_score_kernel(const int* __restrict__ src, const int* __restrict__ dst,
                                  const float* __restrict__ XL, const float* __restrict__ XR,
                                  const float* __restrict__ att,
                                  float* __restrict__ S, float* __restrict__ SMAX,
                                  int ne, int d)
{
    const int t  = blockIdx.x * blockDim.x + threadIdx.x;
    const int e  = t >> 3;         // 8 lanes per edge (aligned groups)
    const int sl = t & 7;
    if (e >= ne) return;
    const int u = src[e], w = dst[e];
    const float4* xl = (const float4*)(XL + (long)u * d);
    const float4* xr = (const float4*)(XR + (long)w * d);
    const float4* a4 = (const float4*)att;
    float acc = 0.0f;
    const int d4 = d >> 2;         // 8 | 16 | 32
    for (int i = sl; i < d4; i += 8) {
        float4 l = xl[i], r = xr[i], a = a4[i];
        float tv;
        tv = l.x + r.x; tv = tv > 0.0f ? tv : NEG_SLOPE * tv; acc += tv * a.x;
        tv = l.y + r.y; tv = tv > 0.0f ? tv : NEG_SLOPE * tv; acc += tv * a.y;
        tv = l.z + r.z; tv = tv > 0.0f ? tv : NEG_SLOPE * tv; acc += tv * a.z;
        tv = l.w + r.w; tv = tv > 0.0f ? tv : NEG_SLOPE * tv; acc += tv * a.w;
    }
    acc += __shfl_xor(acc, 1);
    acc += __shfl_xor(acc, 2);
    acc += __shfl_xor(acc, 4);
    if (sl == 0) {
        S[e] = acc;
        atomicMaxF(&SMAX[w], acc);
    }
}

// pass 2 — exp(s - smax[dst]) (stored in-place into S) + segment sum
__global__ void edge_exp_kernel(const int* __restrict__ dst, float* __restrict__ S,
                                const float* __restrict__ SMAX, float* __restrict__ DEN, int ne)
{
    int e = blockIdx.x * blockDim.x + threadIdx.x;
    if (e >= ne) return;
    const int w = dst[e];
    float m = SMAX[w];
    if (!(fabsf(m) <= 3.0e38f)) m = 0.0f;   // where(isfinite(smax), smax, 0)
    const float v = expf(S[e] - m);
    S[e] = v;
    atomicAdd(&DEN[w], v);
}

// pass 3 — OUT[dst] += alpha * xl[src]; 8 lanes per edge, channel-parallel atomics
__global__ void edge_scatter_kernel(const int* __restrict__ src, const int* __restrict__ dst,
                                    const float* __restrict__ S, const float* __restrict__ DEN,
                                    const float* __restrict__ XL, float* __restrict__ OUT,
                                    int ne, int d)
{
    const int t  = blockIdx.x * blockDim.x + threadIdx.x;
    const int e  = t >> 3;
    const int sl = t & 7;
    if (e >= ne) return;
    const int u = src[e], w = dst[e];
    const float alpha = S[e] / fmaxf(DEN[w], 1e-16f);
    const float4* xl = (const float4*)(XL + (long)u * d);
    float* o = OUT + (long)w * d;
    const int d4 = d >> 2;
    for (int i = sl; i < d4; i += 8) {
        float4 l = xl[i];
        atomicAdd(&o[4 * i + 0], alpha * l.x);
        atomicAdd(&o[4 * i + 1], alpha * l.y);
        atomicAdd(&o[4 * i + 2], alpha * l.z);
        atomicAdd(&o[4 * i + 3], alpha * l.w);
    }
}

// ---------------------------------------------------------------------------
// Pooling: 8 lanes per node (d == 32 -> one float4 per lane)
// ---------------------------------------------------------------------------
__global__ void pool_kernel(const float* __restrict__ H, const int* __restrict__ batch,
                            float* __restrict__ POOL, float* __restrict__ CNT, int n, int d)
{
    const int t  = blockIdx.x * blockDim.x + threadIdx.x;
    const int i  = t >> 3;
    const int sl = t & 7;
    if (i >= n) return;
    const int b = batch[i];
    if (sl == 0) atomicAdd(&CNT[b], 1.0f);
    const float4* h4 = (const float4*)(H + (long)i * d);
    const int d4 = d >> 2;
    for (int c = sl; c < d4; c += 8) {
        float4 h = h4[c];
        float* p = POOL + b * d + 4 * c;
        atomicAdd(p + 0, h.x);
        atomicAdd(p + 1, h.y);
        atomicAdd(p + 2, h.z);
        atomicAdd(p + 3, h.w);
    }
}

__global__ void pool_div_kernel(float* __restrict__ POOL, const float* __restrict__ CNT, int total, int d)
{
    int i = blockIdx.x * blockDim.x + threadIdx.x;
    if (i >= total) return;
    POOL[i] /= fmaxf(CNT[i / d], 1.0f);
}

// ---------------------------------------------------------------------------
// Head (B = 64 graphs, tiny)
// ---------------------------------------------------------------------------
__global__ void conv_info_kernel(const float* __restrict__ X, const int* __restrict__ root,
                                 const float* __restrict__ c1w, const float* __restrict__ c1b,
                                 float* __restrict__ INFOC)
{
    int i = blockIdx.x * blockDim.x + threadIdx.x;
    const int cols = IN_F - 2;
    if (i >= BGRAPH * cols) return;
    const int b = i / cols, c = i % cols;
    const float* x = X + (long)root[b] * IN_F;
    INFOC[i] = c1w[0] * x[c] + c1w[1] * x[c + 1] + c1w[2] * x[c + 2] + c1b[0];
}

__global__ void t2_kernel(const float* __restrict__ c2W, const float* __restrict__ c2b,
                          const float* __restrict__ INFOC, float* __restrict__ T2)
{
    int i = blockIdx.x * blockDim.x + threadIdx.x;
    if (i >= HC1C * BGRAPH) return;
    const int r = i / BGRAPH, b = i % BGRAPH;
    const int cols = IN_F - 2;
    float acc = c2b[r];
    const float* wr = c2W + (long)r * cols;
    const float* in = INFOC + (long)b * cols;
    for (int k = 0; k < cols; ++k) acc += wr[k] * in[k];
    T2[i] = fmaxf(acc, 0.0f);    // T2[r*B + b]
}

__global__ void t3_kernel(const float* __restrict__ c3W, const float* __restrict__ c3b,
                          const float* __restrict__ T2, float* __restrict__ T3)
{
    int i = blockIdx.x * blockDim.x + threadIdx.x;
    if (i >= 64 * BGRAPH) return;
    const int r = i / BGRAPH, b = i % BGRAPH;
    float acc = c3b[r];
    for (int k = 0; k < HC1C; ++k) acc += c3W[r * HC1C + k] * T2[k * BGRAPH + b];
    T3[i] = fmaxf(acc, 0.0f);    // T3[r*B + b]
}

__global__ void tfeat_kernel(const float* __restrict__ T3, float* __restrict__ TF)
{
    int i = blockIdx.x * blockDim.x + threadIdx.x;
    if (i >= BGRAPH * HC2C) return;
    const int b = i / HC2C, c = i % HC2C;
    const float v = 0.5f * (T3[(2 * c) * BGRAPH + b] + T3[(2 * c + 1) * BGRAPH + b]);
    TF[i] = fmaxf(v, 0.0f);
}

__global__ void sx_kernel(const float* __restrict__ H3, const int* __restrict__ root,
                          const float* __restrict__ POOL, const float* __restrict__ linW,
                          const float* __restrict__ linb, float* __restrict__ SX)
{
    int i = blockIdx.x * blockDim.x + threadIdx.x;
    if (i >= BGRAPH * HC2C) return;
    const int b = i / HC2C, oc = i % HC2C;
    const float* hr = H3 + (long)root[b] * HC2C;
    const float* pl = POOL + b * HC2C;
    float acc = linb[oc];
    for (int j = 0; j < HC2C; ++j) acc += hr[j] * linW[j * HC2C + oc];
    for (int j = 0; j < HC2C; ++j) acc += pl[j] * linW[(HC2C + j) * HC2C + oc];
    SX[i] = fmaxf(acc, 0.0f);
}

__global__ void head_kernel(const float* __restrict__ SX, const float* __restrict__ TF,
                            const float* __restrict__ aW1, const float* __restrict__ ab1,
                            const float* __restrict__ aW2,
                            const float* __restrict__ mW1, const float* __restrict__ mb1,
                            const float* __restrict__ mW2, const float* __restrict__ mb2,
                            float* __restrict__ out)
{
    int b = blockIdx.x * blockDim.x + threadIdx.x;
    if (b >= BGRAPH) return;
    float e0[HC2C], e1[HC2C];
    for (int c = 0; c < HC2C; ++c) { e0[c] = SX[b * HC2C + c]; e1[c] = TF[b * HC2C + c]; }
    float w0 = 0.0f, w1 = 0.0f;
    for (int j = 0; j < 16; ++j) {
        float a0 = ab1[j], a1 = ab1[j];
        for (int c = 0; c < HC2C; ++c) { a0 += e0[c] * aW1[c * 16 + j]; a1 += e1[c] * aW1[c * 16 + j]; }
        w0 += tanhf(a0) * aW2[j];
        w1 += tanhf(a1) * aW2[j];
    }
    const float mx = fmaxf(w0, w1);
    float b0 = expf(w0 - mx), b1 = expf(w1 - mx);
    const float inv = 1.0f / (b0 + b1);
    b0 *= inv; b1 *= inv;
    float emb[HC2C];
    for (int c = 0; c < HC2C; ++c) emb[c] = b0 * e0[c] + b1 * e1[c];
    float hv[16];
    for (int j = 0; j < 16; ++j) {
        float a = mb1[j];
        for (int c = 0; c < HC2C; ++c) a += emb[c] * mW1[c * 16 + j];
        hv[j] = tanhf(a);
    }
    float l0 = mb2[0], l1 = mb2[1];
    for (int j = 0; j < 16; ++j) { l0 += hv[j] * mW2[j * 2 + 0]; l1 += hv[j] * mW2[j * 2 + 1]; }
    const float m2 = fmaxf(l0, l1);
    const float p0 = expf(l0 - m2), p1 = expf(l1 - m2);
    const float s = p0 + p1;
    out[b * 2 + 0] = p0 / s;
    out[b * 2 + 1] = p1 / s;
}

// ---------------------------------------------------------------------------
// Host orchestration
// ---------------------------------------------------------------------------
static inline int cdiv(long a, int b) { return (int)((a + b - 1) / b); }

static void convert_input(const float* X, _Float16* Xh, long nrows, int kin, int kin_pad,
                          hipStream_t stream)
{
    const long total4 = nrows * (kin_pad >> 2);
    cvt_f16_pad_kernel<<<cdiv(total4, 256), 256, 0, stream>>>(X, Xh, nrows, kin, kin_pad);
}

static void run_linear(const _Float16* Xh, const float* W, const float* bias, float* Y,
                       int nrows, int kin, int kin_pad, int kout, int relu_act,
                       _Float16* Wp, hipStream_t stream)
{
    const int nkc = kin_pad / 32;
    const int ntn = ((kout + 15) / 16 + 3) & ~3;   // column tiles, rounded up to 4
    pack_w_kernel<<<cdiv(ntn * nkc * 32, 256), 256, 0, stream>>>(W, Wp, kin, kout, ntn, nkc);
    dim3 gw(nrows / 16, ntn / 4);
    wmma_linear_kernel<<<gw, 32, 0, stream>>>(Xh, Wp, bias, Y, nrows, kin_pad, kout, nkc, relu_act);
}

static void run_gat_layer(const float* Hin, int din, int dout,
                          const float* Wl, const float* bl,
                          const float* Wr, const float* br,
                          const float* att, const float* bias,
                          _Float16* Xh, _Float16* Wp,
                          float* XL, float* XR, float* S, float* SMAX, float* DEN,
                          float* Hout, const int* src, const int* dst, hipStream_t stream)
{
    const int din_pad = ((din + 31) / 32) * 32;
    convert_input(Hin, Xh, N_NODES, din, din_pad, stream);
    run_linear(Xh, Wl, bl, XL, N_NODES, din, din_pad, dout, 0, Wp, stream);
    run_linear(Xh, Wr, br, XR, N_NODES, din, din_pad, dout, 0, Wp, stream);

    fill_f32_kernel<<<cdiv(N_NODES, 256), 256, 0, stream>>>(SMAX, -__builtin_inff(), N_NODES);
    fill_f32_kernel<<<cdiv(N_NODES, 256), 256, 0, stream>>>(DEN, 0.0f, N_NODES);
    const long nd = (long)N_NODES * dout;
    init_bias_rows_kernel<<<cdiv(nd, 256), 256, 0, stream>>>(Hout, bias, nd, dout);
    edge_score_kernel<<<cdiv((long)N_EDGES * 8, 256), 256, 0, stream>>>(src, dst, XL, XR, att, S, SMAX, N_EDGES, dout);
    edge_exp_kernel<<<cdiv(N_EDGES, 256), 256, 0, stream>>>(dst, S, SMAX, DEN, N_EDGES);
    edge_scatter_kernel<<<cdiv((long)N_EDGES * 8, 256), 256, 0, stream>>>(src, dst, S, DEN, XL, Hout, N_EDGES, dout);
    relu_ip_kernel<<<cdiv(nd, 256), 256, 0, stream>>>(Hout, nd);
}

extern "C" void kernel_launch(void* const* d_in, const int* in_sizes, int n_in,
                              void* d_out, int out_size, void* d_ws, size_t ws_size,
                              hipStream_t stream)
{
    // ---- inputs (setup_inputs order) ----
    const float* s_x   = (const float*)d_in[0];
    const int*   eidx  = (const int*)d_in[1];
    const int*   batch = (const int*)d_in[2];
    const int*   root  = (const int*)d_in[3];
    // d_in[4..7] = Wq,bq,Wk,bk — mathematically dead (softmax over singleton axis == 1)
    const float* Wv  = (const float*)d_in[8];
    const float* bv  = (const float*)d_in[9];
    const float* Wl1 = (const float*)d_in[10]; const float* bl1 = (const float*)d_in[11];
    const float* Wr1 = (const float*)d_in[12]; const float* br1 = (const float*)d_in[13];
    const float* att1 = (const float*)d_in[14]; const float* bias1 = (const float*)d_in[15];
    const float* Wl2 = (const float*)d_in[16]; const float* bl2 = (const float*)d_in[17];
    const float* Wr2 = (const float*)d_in[18]; const float* br2 = (const float*)d_in[19];
    const float* att2 = (const float*)d_in[20]; const float* bias2 = (const float*)d_in[21];
    const float* Wl3 = (const float*)d_in[22]; const float* bl3 = (const float*)d_in[23];
    const float* Wr3 = (const float*)d_in[24]; const float* br3 = (const float*)d_in[25];
    const float* att3 = (const float*)d_in[26]; const float* bias3 = (const float*)d_in[27];
    const float* c1w = (const float*)d_in[28]; const float* c1b = (const float*)d_in[29];
    const float* c2W = (const float*)d_in[30]; const float* c2b = (const float*)d_in[31];
    const float* c3W = (const float*)d_in[32]; const float* c3b = (const float*)d_in[33];
    const float* linW = (const float*)d_in[34]; const float* linb = (const float*)d_in[35];
    const float* aW1 = (const float*)d_in[36]; const float* ab1 = (const float*)d_in[37];
    const float* aW2 = (const float*)d_in[38];
    const float* mW1 = (const float*)d_in[39]; const float* mb1 = (const float*)d_in[40];
    const float* mW2 = (const float*)d_in[41]; const float* mb2 = (const float*)d_in[42];

    const int* src = eidx;              // edge_index[0]
    const int* dst = eidx + N_EDGES;    // edge_index[1]

    // ---- workspace layout (byte carve, 64B aligned) ----
    char* base = (char*)d_ws;
    size_t off = 0;
    auto carve = [&](size_t bytes) -> void* {
        void* p = base + off;
        off = (off + bytes + 63) & ~(size_t)63;
        return p;
    };
    float* H0   = (float*)carve((size_t)N_NODES * SAC * 4);
    float* H1   = (float*)carve((size_t)N_NODES * HC1C * 4);
    float* XL   = (float*)carve((size_t)N_NODES * HC1C * 4);
    float* XR   = (float*)carve((size_t)N_NODES * HC1C * 4);
    float* S    = (float*)carve((size_t)N_EDGES * 4);
    float* SMAX = (float*)carve((size_t)N_NODES * 4);
    float* DEN  = (float*)carve((size_t)N_NODES * 4);
    _Float16* Xh = (_Float16*)carve((size_t)N_NODES * 352 * 2);  // padded f16 activations
    _Float16* Wp = (_Float16*)carve((size_t)131072 * 2);         // packed f16 weights (reused)
    float* POOL = (float*)carve((size_t)BGRAPH * HC2C * 4);
    float* CNT  = (float*)carve((size_t)BGRAPH * 4);
    float* INFOC= (float*)carve((size_t)BGRAPH * (IN_F - 2) * 4);
    float* T2   = (float*)carve((size_t)HC1C * BGRAPH * 4);
    float* T3   = (float*)carve((size_t)64 * BGRAPH * 4);
    float* TF   = (float*)carve((size_t)BGRAPH * HC2C * 4);
    float* SXB  = (float*)carve((size_t)BGRAPH * HC2C * 4);
    (void)ws_size; (void)in_sizes; (void)n_in; (void)out_size;

    // ---- 1) h = x @ Wv + bv   (singleton-axis softmax collapses attention to v) ----
    convert_input(s_x, Xh, N_NODES, IN_F, IN_F, stream);
    run_linear(Xh, Wv, bv, H0, N_NODES, IN_F, IN_F, SAC, 0, Wp, stream);

    // ---- 2) three GATv2 layers ----
    run_gat_layer(H0, SAC, HC1C,  Wl1, bl1, Wr1, br1, att1, bias1,
                  Xh, Wp, XL, XR, S, SMAX, DEN, H1, src, dst, stream);   // 350 -> 128
    run_gat_layer(H1, HC1C, 64,   Wl2, bl2, Wr2, br2, att2, bias2,
                  Xh, Wp, XL, XR, S, SMAX, DEN, H0, src, dst, stream);   // 128 -> 64 (reuse H0)
    run_gat_layer(H0, 64, HC2C,   Wl3, bl3, Wr3, br3, att3, bias3,
                  Xh, Wp, XL, XR, S, SMAX, DEN, H1, src, dst, stream);   // 64 -> 32 (reuse H1)
    float* H3 = H1;

    // ---- 3) mean pooling per graph ----
    fill_f32_kernel<<<1, 256, 0, stream>>>(POOL, 0.0f, (long)BGRAPH * HC2C);
    fill_f32_kernel<<<1, 256, 0, stream>>>(CNT, 0.0f, (long)BGRAPH);
    pool_kernel<<<cdiv((long)N_NODES * 8, 256), 256, 0, stream>>>(H3, batch, POOL, CNT, N_NODES, HC2C);
    pool_div_kernel<<<cdiv(BGRAPH * HC2C, 256), 256, 0, stream>>>(POOL, CNT, BGRAPH * HC2C, HC2C);

    // ---- 4) head ----
    conv_info_kernel<<<cdiv((long)BGRAPH * (IN_F - 2), 256), 256, 0, stream>>>(s_x, root, c1w, c1b, INFOC);
    t2_kernel<<<cdiv(HC1C * BGRAPH, 256), 256, 0, stream>>>(c2W, c2b, INFOC, T2);
    t3_kernel<<<cdiv(64 * BGRAPH, 256), 256, 0, stream>>>(c3W, c3b, T2, T3);
    tfeat_kernel<<<cdiv(BGRAPH * HC2C, 256), 256, 0, stream>>>(T3, TF);
    sx_kernel<<<cdiv(BGRAPH * HC2C, 256), 256, 0, stream>>>(H3, root, POOL, linW, linb, SXB);
    head_kernel<<<cdiv(BGRAPH, 64), 64, 0, stream>>>(SXB, TF, aW1, ab1, aW2,
                                                     mW1, mb1, mW2, mb2, (float*)d_out);
}
